// MultiheadKANAttention_67637144978234
// MI455X (gfx1250) — compile-verified, matrix operand order checked by reading, not run on hardware
//
#include <hip/hip_runtime.h>
#include <stdint.h>

// ---------------------------------------------------------------------------
// MultiheadKANAttention for MI455X (gfx1250), bf16 WMMA pipeline.
// B=4, S=2048, D_MODEL=512, HEADS=8, HEAD_DIM=64, GRID=8
// ---------------------------------------------------------------------------

typedef __bf16  v16bf __attribute__((ext_vector_type(16)));
typedef float   v8f   __attribute__((ext_vector_type(8)));

#define NB      4
#define NS      2048
#define DM      512
#define NH      8
#define HD      64
#define NGRID   8
#define NTOK    (NB * NS)          // 8192 tokens
#define NFEAT   (DM * NGRID * 2)   // 8192 sin/cos features
#define NQKV    (3 * DM)           // 1536
#define HEAD_ELEMS 131072          // per-(b,h) fragment elems for Q/K/V

// ---- bf16 <-> f32 helpers (round-to-nearest-even) -------------------------
__device__ __forceinline__ uint16_t f2bf(float f) {
  union { float f; uint32_t u; } x; x.f = f;
  uint32_t r = x.u + 0x7FFFu + ((x.u >> 16) & 1u);
  return (uint16_t)(r >> 16);
}
__device__ __forceinline__ float bf2f(uint16_t u) {
  union { uint32_t u; float f; } x; x.u = ((uint32_t)u) << 16;
  return x.f;
}

// ---- WMMA 16x16x32 bf16 fragment address helper ---------------------------
// Per CDNA5 ISA 7.12.2 (16-bit A-matrix 16x32 / B-matrix 32x16):
//   lane(0..31) = (row_or_col & 15) + 16 * ((k_local >> 3) & 1)
//   elem(0..15) = (k_local & 7) + ((k_local & 16) ? 8 : 0)
// Fragments stored as [tile][lane][elem] -> each lane loads 16 contiguous bf16.
__device__ __forceinline__ size_t frag_addr(int rc, int k, int nkt) {
  int rt = rc >> 4, m = rc & 15;
  int kt = k >> 5, kl = k & 31;
  int lane = m + (((kl >> 3) & 1) << 4);
  int e = (kl & 7) + ((kl & 16) ? 8 : 0);
  return ((size_t)(rt * (size_t)nkt + kt) * 32 + (size_t)lane) * 16 + (size_t)e;
}

// ===========================================================================
// K1: KAN weight pack:  W[j, k*16 + 2l + t] -> bf16 B-fragments (K=8192)
// ===========================================================================
__global__ void __launch_bounds__(256)
kan_weight_pack_kernel(const float* __restrict__ amp,
                       const float* __restrict__ phase,
                       uint16_t* __restrict__ Wf) {
  int idx = blockIdx.x * blockDim.x + threadIdx.x;   // j*512 + k
  int k = idx & (DM - 1);
  int j = idx >> 9;
#pragma unroll
  for (int l = 0; l < NGRID; ++l) {
    float ph = phase[k * NGRID + l];
    float a  = amp[((size_t)j * DM + k) * NGRID + l];
    int f = k * 16 + l * 2;
    Wf[frag_addr(j, f,     NFEAT / 32)] = f2bf(a * __cosf(ph));  // pairs sin(ang)
    Wf[frag_addr(j, f + 1, NFEAT / 32)] = f2bf(a * __sinf(ph));  // pairs cos(ang)
  }
}

// ===========================================================================
// K2: feature generation: F[n, f] = sin/cos(q[n,k]*freq[l]) -> bf16 A-frags
// ===========================================================================
__global__ void __launch_bounds__(256)
kan_feature_kernel(const float* __restrict__ q,
                   const float* __restrict__ freq,
                   uint16_t* __restrict__ Ff) {
  int idx = blockIdx.x * blockDim.x + threadIdx.x;   // n*512 + k
  int k = idx & (DM - 1);
  int n = idx >> 9;
  float x = q[idx];
#pragma unroll
  for (int l = 0; l < NGRID; ++l) {
    float ang = x * freq[l];
    int f = k * 16 + l * 2;
    Ff[frag_addr(n, f,     NFEAT / 32)] = f2bf(__sinf(ang));
    Ff[frag_addr(n, f + 1, NFEAT / 32)] = f2bf(__cosf(ang));
  }
}

// ===========================================================================
// K3/K7: generic bf16 WMMA GEMM on pre-swizzled fragments.
//   rows x ncols, K = nkt*32.  Block tile 128x128, 256 thr = 8 waves (2x4),
//   wave tile 64x32 -> 8 C fragments.  Streams fragments straight from L2.
// ===========================================================================
template <bool BF16OUT>
__global__ void __launch_bounds__(256)
wmma_gemm_kernel(const uint16_t* __restrict__ Af,
                 const uint16_t* __restrict__ Bf,
                 const float* __restrict__ bias,
                 float* __restrict__ outF,
                 uint16_t* __restrict__ outH,
                 int nkt, int ncols) {
  const int lane = threadIdx.x & 31;
  const int w    = threadIdx.x >> 5;
  const int wr   = w >> 2, wc = w & 3;
  const int rowTile0 = blockIdx.y * 8 + wr * 4;   // 4 row fragments
  const int colTile0 = blockIdx.x * 8 + wc * 2;   // 2 col fragments

  v8f acc[4][2] = {};

  for (int kt = 0; kt < nkt; ++kt) {
    v16bf a[4], b[2];
#pragma unroll
    for (int i = 0; i < 4; ++i)
      a[i] = *(const v16bf*)(Af + ((size_t)(rowTile0 + i) * nkt + kt) * 512 + lane * 16);
#pragma unroll
    for (int jf = 0; jf < 2; ++jf)
      b[jf] = *(const v16bf*)(Bf + ((size_t)(colTile0 + jf) * nkt + kt) * 512 + lane * 16);
    if (kt + 1 < nkt) {   // L2 prefetch of next K-step (global_prefetch_b8)
      __builtin_prefetch(Af + ((size_t)rowTile0 * nkt + kt + 1) * 512 + lane * 16, 0, 1);
      __builtin_prefetch(Bf + ((size_t)colTile0 * nkt + kt + 1) * 512 + lane * 16, 0, 1);
    }
#pragma unroll
    for (int i = 0; i < 4; ++i)
#pragma unroll
      for (int jf = 0; jf < 2; ++jf)
        acc[i][jf] = __builtin_amdgcn_wmma_f32_16x16x32_bf16(
            false, a[i], false, b[jf], (short)0, acc[i][jf], false, false);
  }

  // Epilogue: C layout — lanes 0-15: M=r, lanes 16-31: M=r+8; N = lane&15.
  const int hi = lane >> 4, nc = lane & 15;
#pragma unroll
  for (int jf = 0; jf < 2; ++jf) {
    int jcol = (colTile0 + jf) * 16 + nc;
    float bb = bias[jcol];
#pragma unroll
    for (int i = 0; i < 4; ++i) {
#pragma unroll
      for (int r = 0; r < 8; ++r) {
        int row = (rowTile0 + i) * 16 + r + 8 * hi;
        float v = acc[i][jf][r] + bb;
        if (BF16OUT) outH[(size_t)row * ncols + jcol] = f2bf(v);
        else         outF[(size_t)row * ncols + jcol] = v;
      }
    }
  }
}

// ===========================================================================
// K4: RoPE + per-head fragment pack.
//   qkv[n, h*192 + {q:0-63, k:64-127, v:128-191}] (bf16) ->
//   Qf (A-frag rows=s,K=d), Kf (B-frag cols=s,K=d), Vf (B-frag K=s32,cols=d)
// ===========================================================================
__global__ void __launch_bounds__(256)
rope_pack_kernel(const uint16_t* __restrict__ qkv,
                 uint16_t* __restrict__ Qf,
                 uint16_t* __restrict__ Kf,
                 uint16_t* __restrict__ Vf) {
  int idx = blockIdx.x * blockDim.x + threadIdx.x;   // ((b*8+h)*2048+s)*64+d
  int d = idx & 63;
  int s = (idx >> 6) & (NS - 1);
  int h = (idx >> 17) & 7;
  int b = idx >> 20;
  const uint16_t* base = qkv + ((size_t)(b * NS + s)) * NQKV + h * (3 * HD);

  int i = (d < 32) ? d : d - 32;
  // inv_freq = 10000^(-i/32) ; log2(10000) = 13.2877123795...
  float ang = (float)s * exp2f(-(float)i * (13.287712379549449f / 32.0f));
  float cw = __cosf(ang), sw = __sinf(ang);

  auto rot = [&](int off) -> float {
    if (d < 32) {                                   // x1 = x_even*cos + rh(x_odd)*sin
      float xe = bf2f(base[off + 2 * i]);
      if (i < 16) return xe * cw - bf2f(base[off + 2 * (i + 16) + 1]) * sw;
      else        return xe * cw + bf2f(base[off + 2 * (i - 16) + 1]) * sw;
    } else {                                        // x2 = x_odd*cos + rh(x_even)*sin
      float xo = bf2f(base[off + 2 * i + 1]);
      if (i < 16) return xo * cw - bf2f(base[off + 2 * (i + 16)]) * sw;
      else        return xo * cw + bf2f(base[off + 2 * (i - 16)]) * sw;
    }
  };

  size_t hb = (size_t)(b * NH + h) * HEAD_ELEMS;
  Qf[hb + frag_addr(s, d, HD / 32)] = f2bf(rot(0));
  Kf[hb + frag_addr(s, d, HD / 32)] = f2bf(rot(HD));

  // V B-fragment for PV: K-dim = key position (32/tile), N = d (4 tiles of 16)
  int sl = s & 31, lane = (d & 15) + (((sl >> 3) & 1) << 4);
  int e = (sl & 7) + ((sl & 16) ? 8 : 0);
  Vf[hb + ((size_t)((s >> 5) * 4 + (d >> 4)) * 32 + lane) * 16 + e] =
      base[2 * HD + d];   // raw bf16 bits, no rope on V
}

// ===========================================================================
// K5: causal flash attention, bf16 WMMA, online softmax.
//   Block: 4 waves x 16 queries = 64 queries.  Loop key tiles of 32.
// ===========================================================================
__global__ void __launch_bounds__(128)
flash_attn_kernel(const uint16_t* __restrict__ Qf,
                  const uint16_t* __restrict__ Kf,
                  const uint16_t* __restrict__ Vf,
                  float* __restrict__ ctx) {
  __shared__ __align__(32) uint16_t pbuf[4][512];   // per-wave P staging (16x32 bf16)

  const int lane = threadIdx.x & 31;
  const int wave = threadIdx.x >> 5;
  const int qt64 = blockIdx.x;              // 0..31
  const int h    = blockIdx.y;              // 0..7
  const int b    = blockIdx.z;              // 0..3
  const size_t hb = (size_t)(b * NH + h) * HEAD_ELEMS;
  const int qTile = qt64 * 4 + wave;        // 16-query tile
  const int qbase = qTile * 16;
  const int hi = lane >> 4, nc = lane & 15;

  v16bf qa[2];
#pragma unroll
  for (int kt = 0; kt < 2; ++kt)
    qa[kt] = *(const v16bf*)(Qf + hb + ((size_t)(qTile * 2 + kt) * 32 + lane) * 16);

  v8f accO[4] = {};
  float mr[8], lr[8];
#pragma unroll
  for (int r = 0; r < 8; ++r) { mr[r] = -1e30f; lr[r] = 0.0f; }

  const int nkt32 = qt64 * 2 + 2;           // uniform across the block
  for (int kt32 = 0; kt32 < nkt32; ++kt32) {
    // --- scores: S = Q (16x64) @ K^T (64x32) -> two 16x16 C fragments ---
    v8f sc[2] = {};
#pragma unroll
    for (int half = 0; half < 2; ++half) {
      int keyT16 = kt32 * 2 + half;
#pragma unroll
      for (int kt = 0; kt < 2; ++kt) {
        v16bf kb = *(const v16bf*)(Kf + hb + ((size_t)(keyT16 * 2 + kt) * 32 + lane) * 16);
        sc[half] = __builtin_amdgcn_wmma_f32_16x16x32_bf16(
            false, qa[kt], false, kb, (short)0, sc[half], false, false);
      }
    }
    // --- scale, causal mask, online softmax (row = r + 8*hi across 16 lanes) ---
#pragma unroll
    for (int r = 0; r < 8; ++r) {
      int m   = r + 8 * hi;
      int row = qbase + m;
      int k0  = kt32 * 32 + nc, k1 = k0 + 16;
      float s0 = sc[0][r] * 0.125f;           // 1/sqrt(64)
      float s1 = sc[1][r] * 0.125f;
      if (k0 > row) s0 = -1e9f;
      if (k1 > row) s1 = -1e9f;
      float mx = fmaxf(s0, s1);
#pragma unroll
      for (int off = 8; off >= 1; off >>= 1)
        mx = fmaxf(mx, __shfl_xor(mx, off, 32));   // 16-lane group reduce
      float mnew  = fmaxf(mr[r], mx);
      float alpha = __expf(mr[r] - mnew);
      float p0 = __expf(s0 - mnew);
      float p1 = __expf(s1 - mnew);
      float rs = p0 + p1;
#pragma unroll
      for (int off = 8; off >= 1; off >>= 1)
        rs += __shfl_xor(rs, off, 32);
      lr[r] = lr[r] * alpha + rs;
      mr[r] = mnew;
#pragma unroll
      for (int dt = 0; dt < 4; ++dt) accO[dt][r] *= alpha;
      // stage P into LDS in A-fragment layout (16 rows x 32 keys)
      int c0 = nc, c1 = nc + 16;
      pbuf[wave][(m + (((c0 >> 3) & 1) << 4)) * 16 + (c0 & 7) + ((c0 & 16) ? 8 : 0)] = f2bf(p0);
      pbuf[wave][(m + (((c1 >> 3) & 1) << 4)) * 16 + (c1 & 7) + ((c1 & 16) ? 8 : 0)] = f2bf(p1);
    }
    __syncthreads();
    v16bf pa = *(const v16bf*)&pbuf[wave][lane * 16];
#pragma unroll
    for (int dt = 0; dt < 4; ++dt) {
      v16bf vb = *(const v16bf*)(Vf + hb + ((size_t)(kt32 * 4 + dt) * 32 + lane) * 16);
      accO[dt] = __builtin_amdgcn_wmma_f32_16x16x32_bf16(
          false, pa, false, vb, (short)0, accO[dt], false, false);
    }
    __syncthreads();
  }

  // --- epilogue: ctx[b, s, h*64 + d] = O / l ---
#pragma unroll
  for (int dt = 0; dt < 4; ++dt) {
    int d = dt * 16 + nc;
#pragma unroll
    for (int r = 0; r < 8; ++r) {
      int row = qbase + r + 8 * hi;
      ctx[((size_t)(b * NS + row)) * DM + h * HD + d] = accO[dt][r] / lr[r];
    }
  }
}

// ===========================================================================
// K6a/K6b: pack ctx (A-frags) and out_w (B-frags) for the projection GEMM
// ===========================================================================
__global__ void __launch_bounds__(256)
ctx_pack_kernel(const float* __restrict__ ctx, uint16_t* __restrict__ Cf) {
  int idx = blockIdx.x * blockDim.x + threadIdx.x;   // n*512 + d
  int d = idx & (DM - 1), n = idx >> 9;
  Cf[frag_addr(n, d, DM / 32)] = f2bf(ctx[idx]);
}
__global__ void __launch_bounds__(256)
outw_pack_kernel(const float* __restrict__ out_w, uint16_t* __restrict__ Of) {
  int idx = blockIdx.x * blockDim.x + threadIdx.x;   // o*512 + d
  int d = idx & (DM - 1), o = idx >> 9;
  Of[frag_addr(o, d, DM / 32)] = f2bf(out_w[idx]);
}

// ===========================================================================
// Launch
// ===========================================================================
extern "C" void kernel_launch(void* const* d_in, const int* in_sizes, int n_in,
                              void* d_out, int out_size, void* d_ws, size_t ws_size,
                              hipStream_t stream) {
  const float* q     = (const float*)d_in[0];
  // d_in[1] = mask (causal, implemented analytically)
  const float* freq  = (const float*)d_in[2];
  const float* phase = (const float*)d_in[3];
  const float* amp   = (const float*)d_in[4];
  const float* kbias = (const float*)d_in[5];
  const float* outw  = (const float*)d_in[6];
  const float* outb  = (const float*)d_in[7];
  float* out = (float*)d_out;

  char* ws = (char*)d_ws;
  const size_t szF = (size_t)NTOK * NFEAT * 2;   // 134,217,728 B  feature frags
  const size_t szW = (size_t)NFEAT * NQKV * 2;   //  25,165,824 B  KAN weight frags
  uint16_t* Ff  = (uint16_t*)ws;
  uint16_t* Wkf = (uint16_t*)(ws + szF);
  uint16_t* qkv = (uint16_t*)(ws + szF + szW);   //  25,165,824 B  qkv bf16

  // F is dead after the KAN GEMM -> overlay attention/projection buffers on it.
  const size_t szQKVfrag = (size_t)NB * NH * HEAD_ELEMS * 2;   // 8,388,608 B each
  uint16_t* Qf  = (uint16_t*)ws;
  uint16_t* Kf  = (uint16_t*)(ws + szQKVfrag);
  uint16_t* Vf  = (uint16_t*)(ws + 2 * szQKVfrag);
  float*    ctx = (float*)   (ws + 3 * szQKVfrag);             // 16,777,216 B
  uint16_t* Cf  = (uint16_t*)(ws + 3 * szQKVfrag + (size_t)NTOK * DM * 4);
  uint16_t* Owf = (uint16_t*)(ws + 3 * szQKVfrag + (size_t)NTOK * DM * 4
                                 + (size_t)NTOK * DM * 2);

  // 1) pack KAN weights + generate features (bf16, WMMA-fragment layout)
  kan_weight_pack_kernel<<<(NQKV * DM) / 256, 256, 0, stream>>>(amp, phase, Wkf);
  kan_feature_kernel<<<(NTOK * DM) / 256, 256, 0, stream>>>(q, freq, Ff);

  // 2) KAN GEMM: 8192 x 1536, K = 8192  (the 206-GFLOP hot spot)
  wmma_gemm_kernel<true><<<dim3(NQKV / 128, NTOK / 128), 256, 0, stream>>>(
      Ff, Wkf, kbias, nullptr, qkv, NFEAT / 32, NQKV);

  // 3) RoPE + per-head fragment pack
  rope_pack_kernel<<<(NB * NH * NS * HD) / 256, 256, 0, stream>>>(qkv, Qf, Kf, Vf);

  // 4) causal flash attention
  flash_attn_kernel<<<dim3(NS / 64, NH, NB), 128, 0, stream>>>(Qf, Kf, Vf, ctx);

  // 5) output projection: 8192 x 512, K = 512
  ctx_pack_kernel<<<(NTOK * DM) / 256, 256, 0, stream>>>(ctx, Cf);
  outw_pack_kernel<<<(DM * DM) / 256, 256, 0, stream>>>(outw, Owf);
  wmma_gemm_kernel<false><<<dim3(DM / 128, NTOK / 128), 256, 0, stream>>>(
      Cf, Owf, outb, out, nullptr, DM / 32, DM);
}
